// MLPDecoder_87290915324173
// MI455X (gfx1250) — compile-verified, matrix operand order checked by reading, not run on hardware
//
#include <hip/hip_runtime.h>

typedef __attribute__((ext_vector_type(16))) _Float16 v16h;
typedef __attribute__((ext_vector_type(8)))  float    v8f;

#define Bsz   8
#define Nn    100
#define Tt    16
#define Dd    4
#define Hh    64
#define Ee    9900   // N*(N-1)
#define TOUT  15     // only first T-1 timesteps are emitted
#define NTHR  512    // 16 wave32s -> fills one WGP

// LDS row stride padded to 65 floats to avoid 64-bank conflicts on per-sender reads.
#define LSTR 65

__global__ __launch_bounds__(NTHR) void nri_decoder_step(
    const float* __restrict__ inputs,     // (B,N,T,D)
    const float* __restrict__ rel_type,   // (B,E,K=2)
    const float* __restrict__ msg_fc1_w,  // (2,H,8)
    const float* __restrict__ msg_fc1_b,  // (2,H)
    const float* __restrict__ msg_fc2_w,  // (2,H,H)
    const float* __restrict__ msg_fc2_b,  // (2,H)
    const float* __restrict__ out_fc1_w,  // (H,D+H)
    const float* __restrict__ out_fc1_b,  // (H)
    const float* __restrict__ out_fc2_w,  // (H,H)
    const float* __restrict__ out_fc2_b,  // (H)
    const float* __restrict__ out_fc3_w,  // (D,H)
    const float* __restrict__ out_fc3_b,  // (D)
    float* __restrict__ out)              // (B,N,T-1,D)
{
  __shared__ float sX[Nn * Dd];          // node states x[b,t,:,:]
  __shared__ float sW1[Hh * 8];          // msg_fc1_w[1]
  __shared__ float sB1[Hh];              // msg_fc1_b[1]
  __shared__ float sUS[Nn * LSTR];       // u_send (edge phase) / p1 (node phase)
  __shared__ float sAGG[Nn * LSTR];      // agg     (edge phase) / p2 (node phase)
  __shared__ float sRT[NTHR / 32][16];   // per-wave rel_type tile

  const int blk  = blockIdx.x;
  const int b    = blk / TOUT;
  const int t    = blk % TOUT;
  const int tid  = threadIdx.x;
  const int lane = tid & 31;
  const int wv   = tid >> 5;             // wave id (0..15)
  const int g    = lane >> 4;            // half-wave group
  const int ln   = lane & 15;

  const float* w1  = msg_fc1_w + Hh * 8;     // k=1 slice
  const float* b1  = msg_fc1_b + Hh;
  const float* w2  = msg_fc2_w + Hh * Hh;
  const float* b2  = msg_fc2_b + Hh;
  const float* rt1 = rel_type + (size_t)b * Ee * 2;

  // ---- stage x, W1, b1 into LDS ----
  for (int i = tid; i < Nn * Dd; i += NTHR) {
    int n = i >> 2, d = i & 3;
    sX[i] = inputs[(((size_t)b * Nn + n) * Tt + t) * Dd + d];
  }
  for (int i = tid; i < Hh * 8; i += NTHR) sW1[i] = w1[i];
  for (int i = tid; i < Hh;     i += NTHR) sB1[i] = b1[i];
  __syncthreads();

  // ---- u_send[n][h] = sum_d W1[h][d] * x[n][d] ----
  for (int i = tid; i < Nn * Hh; i += NTHR) {
    int n = i >> 6, h = i & 63;
    float a = 0.f;
#pragma unroll
    for (int d = 0; d < 4; ++d) a += sW1[h * 8 + d] * sX[n * 4 + d];
    sUS[n * LSTR + h] = a;
  }
  __syncthreads();

  // ---- resident B-fragments of W2^T: B[k][n] = W2[nglob][k] ----
  // 16-bit operand layout: lane (col n, half g), element j <-> K = 32*ks + 8*g + (j<8 ? j : j+8)
  v16h bf[4][2];
#pragma unroll
  for (int nt = 0; nt < 4; ++nt) {
    int nglob = nt * 16 + ln;
#pragma unroll
    for (int ks = 0; ks < 2; ++ks) {
#pragma unroll
      for (int j = 0; j < 16; ++j) {
        int kk = 32 * ks + 8 * g + (j < 8 ? j : j + 8);
        bf[nt][ks][j] = (_Float16)w2[nglob * Hh + kk];
      }
    }
  }
  float b2c[4];
#pragma unroll
  for (int nt = 0; nt < 4; ++nt) b2c[nt] = b2[nt * 16 + ln];

  // ---- edge phase: wave wv owns receivers r = wv, wv+16, ... ----
  for (int r = wv; r < Nn; r += NTHR / 32) {
    // receiver part of h1 (+bias), cached per lane for this lane's 32 K slots
    float urb[2][16];
    float xr0 = sX[r*4+0], xr1 = sX[r*4+1], xr2 = sX[r*4+2], xr3 = sX[r*4+3];
#pragma unroll
    for (int ks = 0; ks < 2; ++ks)
#pragma unroll
      for (int j = 0; j < 16; ++j) {
        int kk = 32 * ks + 8 * g + (j < 8 ? j : j + 8);
        urb[ks][j] = sB1[kk] + sW1[kk*8+4]*xr0 + sW1[kk*8+5]*xr1
                   + sW1[kk*8+6]*xr2 + sW1[kk*8+7]*xr3;
      }

    float acc[4] = {0.f, 0.f, 0.f, 0.f};

    for (int tile = 0; tile < 7; ++tile) {
      // sender for row m = ln of this tile (senders = 0..99 excluding r)
      int  idx   = tile * 16 + ln;
      int  s     = (idx < r) ? idx : idx + 1;
      if (s > Nn - 1) s = Nn - 1;          // clamp padded rows
      bool valid = (idx < Nn - 1);

      if (g == 0) {                         // stash rel_type[.,e,1] for the 16 rows
        float rt = 0.f;
        if (valid) {
          int e = s * (Nn - 1) + ((r < s) ? r : r - 1);
          rt = rt1[e * 2 + 1];
        }
        sRT[wv][ln] = rt;
      }
      __builtin_amdgcn_wave_barrier();

      // A fragments: h1 = relu(u_send[s] + urb), packed into 16x32 f16 layout
      v16h af[2];
#pragma unroll
      for (int ks = 0; ks < 2; ++ks)
#pragma unroll
        for (int j = 0; j < 16; ++j) {
          int kk = 32 * ks + 8 * g + (j < 8 ? j : j + 8);
          float v = sUS[s * LSTR + kk] + urb[ks][j];
          af[ks][j] = (_Float16)(v > 0.f ? v : 0.f);
        }

      float rtrow[8];
#pragma unroll
      for (int v = 0; v < 8; ++v) rtrow[v] = sRT[wv][v + 8 * g];

#pragma unroll
      for (int nt = 0; nt < 4; ++nt) {
        v8f c = {};
        c = __builtin_amdgcn_wmma_f32_16x16x32_f16(false, af[0], false, bf[nt][0],
                                                   (short)0, c, false, false);
        c = __builtin_amdgcn_wmma_f32_16x16x32_f16(false, af[1], false, bf[nt][1],
                                                   (short)0, c, false, false);
        // h2 = relu(c + b2); message = rt * h2; reduce over the 8 rows this lane holds
        float sum = 0.f;
#pragma unroll
        for (int v = 0; v < 8; ++v) {
          float h2 = c[v] + b2c[nt];
          h2 = h2 > 0.f ? h2 : 0.f;
          sum += rtrow[v] * h2;
        }
        acc[nt] += sum;
      }
      __builtin_amdgcn_wave_barrier();
    }

    // combine half-wave partials (rows 0-7 vs 8-15) and commit agg[r]
#pragma unroll
    for (int nt = 0; nt < 4; ++nt) {
      float tot = acc[nt] + __shfl_xor(acc[nt], 16, 32);
      if (g == 0) sAGG[r * LSTR + nt * 16 + ln] = tot;
    }
  }
  __syncthreads();

  // ---- node MLP: p1 = relu([x, agg] @ W1o^T + b) -> sUS ----
  for (int i = tid; i < Nn * Hh; i += NTHR) {
    int n = i >> 6, h = i & 63;
    const float* wr = out_fc1_w + h * (Dd + Hh);
    float a = out_fc1_b[h];
#pragma unroll
    for (int d = 0; d < 4; ++d) a += wr[d] * sX[n * 4 + d];
#pragma unroll 8
    for (int k = 0; k < Hh; ++k) a += wr[4 + k] * sAGG[n * LSTR + k];
    sUS[n * LSTR + h] = a > 0.f ? a : 0.f;
  }
  __syncthreads();

  // ---- p2 = relu(p1 @ W2o^T + b) -> sAGG ----
  for (int i = tid; i < Nn * Hh; i += NTHR) {
    int n = i >> 6, h = i & 63;
    const float* wr = out_fc2_w + h * Hh;
    float a = out_fc2_b[h];
#pragma unroll 8
    for (int k = 0; k < Hh; ++k) a += wr[k] * sUS[n * LSTR + k];
    sAGG[n * LSTR + h] = a > 0.f ? a : 0.f;
  }
  __syncthreads();

  // ---- p3 + residual, store transposed (B,N,T-1,D) ----
  for (int i = tid; i < Nn * Dd; i += NTHR) {
    int n = i >> 2, d = i & 3;
    const float* wr = out_fc3_w + d * Hh;
    float a = out_fc3_b[d];
#pragma unroll 8
    for (int k = 0; k < Hh; ++k) a += wr[k] * sAGG[n * LSTR + k];
    out[(((size_t)b * Nn + n) * TOUT + t) * Dd + d] = sX[i] + a;
  }
}

extern "C" void kernel_launch(void* const* d_in, const int* in_sizes, int n_in,
                              void* d_out, int out_size, void* d_ws, size_t ws_size,
                              hipStream_t stream) {
  const float* inputs    = (const float*)d_in[0];
  const float* rel_type  = (const float*)d_in[1];
  // d_in[2] rel_rec, d_in[3] rel_send: one-hot matrices replaced by index math
  const float* msg_fc1_w = (const float*)d_in[4];
  const float* msg_fc1_b = (const float*)d_in[5];
  const float* msg_fc2_w = (const float*)d_in[6];
  const float* msg_fc2_b = (const float*)d_in[7];
  const float* out_fc1_w = (const float*)d_in[8];
  const float* out_fc1_b = (const float*)d_in[9];
  const float* out_fc2_w = (const float*)d_in[10];
  const float* out_fc2_b = (const float*)d_in[11];
  const float* out_fc3_w = (const float*)d_in[12];
  const float* out_fc3_b = (const float*)d_in[13];
  // d_in[14] pred_steps == 1 (single step unrolled)
  float* outp = (float*)d_out;

  nri_decoder_step<<<dim3(Bsz * TOUT), dim3(NTHR), 0, stream>>>(
      inputs, rel_type, msg_fc1_w, msg_fc1_b, msg_fc2_w, msg_fc2_b,
      out_fc1_w, out_fc1_b, out_fc2_w, out_fc2_b, out_fc3_w, out_fc3_b, outp);
}
